// UncertaintyAwareEarlyAttentionVariationalBatchGAT_5901285065233
// MI455X (gfx1250) — matrix operand order repaired and checked by reading, not compiled, not run on hardware
//
#include <hip/hip_runtime.h>
#include <hip/hip_bf16.h>
#include <math.h>

// ---------------------------------------------------------------------------
// UncertaintyAwareEarlyAttentionVariationalBatchGAT for MI455X (gfx1250).
// All GEMMs use v_wmma_f32_16x16x32_f16 (wave32, one wave per 16x16 C tile).
// Working set ~55MB -> fully L2-resident (192MB). Per-sample attention is
// never materialized: softmax + var/mean filter fused per row across S.
// Layer-1 attention collapses to row n=N-1 (only node read by the output).
// av0's K loop is fully unrolled (8 WMMAs, deep load clauses); gemm1 is
// unrolled x4 to balance pipelining against VGPR pressure.
// ---------------------------------------------------------------------------

typedef __attribute__((ext_vector_type(16))) _Float16 v16h;
typedef __attribute__((ext_vector_type(8)))  float    v8f;

#define SAMP 4
#define BATCH 16
#define NN 256
#define NH0 8
#define F0OUT 64
#define K0PAD 96      // 66 padded to 3*32
#define F1IN 512
#define F1OUT 16
#define A_LIMIT 0.5f
#define A_FILTERED 0.01f
#define NEGINF (-1e9f)

// ----------------------------- threefry2x32 --------------------------------
__device__ __forceinline__ unsigned rotl32(unsigned x, unsigned n) {
    return (x << n) | (x >> (32u - n));
}

__device__ __forceinline__ void threefry2x32(unsigned k0, unsigned k1,
                                             unsigned c0, unsigned c1,
                                             unsigned& o0, unsigned& o1) {
    unsigned ks2 = k0 ^ k1 ^ 0x1BD11BDAu;
    unsigned x0 = c0 + k0, x1 = c1 + k1;
    const unsigned rA0 = 13, rA1 = 15, rA2 = 26, rA3 = 6;
    const unsigned rB0 = 17, rB1 = 29, rB2 = 16, rB3 = 24;
#define TF_R(rr) { x0 += x1; x1 = rotl32(x1, rr); x1 ^= x0; }
    TF_R(rA0) TF_R(rA1) TF_R(rA2) TF_R(rA3)
    x0 += k1;  x1 += ks2 + 1u;
    TF_R(rB0) TF_R(rB1) TF_R(rB2) TF_R(rB3)
    x0 += ks2; x1 += k0 + 2u;
    TF_R(rA0) TF_R(rA1) TF_R(rA2) TF_R(rA3)
    x0 += k0;  x1 += k1 + 3u;
    TF_R(rB0) TF_R(rB1) TF_R(rB2) TF_R(rB3)
    x0 += k1;  x1 += ks2 + 4u;
    TF_R(rA0) TF_R(rA1) TF_R(rA2) TF_R(rA3)
    x0 += ks2; x1 += k0 + 5u;
#undef TF_R
    o0 = x0; o1 = x1;
}

__device__ __forceinline__ float normal_sample(int foldIdx, unsigned j) {
    unsigned fk0, fk1, r0, r1;
    threefry2x32(0u, 42u, 0u, (unsigned)foldIdx, fk0, fk1);   // fold_in(key(42), i)
    threefry2x32(fk0, fk1, 0u, j, r0, r1);                    // counter-mode bits
    float u = ((float)(r0 >> 8) * 0x1.0p-24f) * 2.0f - 1.0f;  // (-1, 1)
    u = fminf(fmaxf(u, -0.999999f), 0.999999f);
    return 1.41421356237f * erfinvf(u);                        // sqrt(2)*erfinv(u)
}

// ------------------------- WMMA tile load helpers ---------------------------
// A tile 16x32 f16, row-major, leading dim lda (elements).
// Layout: lane<16 -> M=lane, K = 0..15; lane>=16 -> K = 16..31 (matches load_bt16).
__device__ __forceinline__ v16h load_a16(const _Float16* A, int lda, int lane) {
    int r = lane & 15, hf = lane >> 4;
    const _Float16* p0 = A + r * lda + 16 * hf;
    const _Float16* p1 = A + r * lda + 8 + 16 * hf;
    v16h a;
#pragma unroll
    for (int e = 0; e < 8; ++e) { a[e] = p0[e]; a[e + 8] = p1[e]; }
    return a;
}

// B tile 32x16 f16 from a *transposed* (n-major over K) buffer Bt[n][k], ldb over k.
// Contiguous k order (as load_a16): lanes 0-15 -> N=lane, K=0..15; lanes 16-31 -> N=lane-16, K=16..31.
__device__ __forceinline__ v16h load_bt16(const _Float16* Bt, int ldb, int lane) {
    int n = lane & 15, hf = lane >> 4;
    const _Float16* p = Bt + n * ldb + 16 * hf;
    v16h b;
#pragma unroll
    for (int e = 0; e < 16; ++e) b[e] = p[e];
    return b;
}

// ------------------------------- RNG draw -----------------------------------
// out[s*nPer + p] = mu[p] + std[p] * eps(fold, s*nPer+p)
__global__ void draw_kernel(const float* __restrict__ mu, const float* __restrict__ sd,
                            float* __restrict__ out, int nPer, int total, int foldIdx) {
    int j = blockIdx.x * blockDim.x + threadIdx.x;
    if (j >= total) return;
    int p = j % nPer;
    out[j] = mu[p] + sd[p] * normal_sample(foldIdx, (unsigned)j);
}

// pack w0 [S,H,66,64] f32 -> w0T [S,H,64,96] f16 (K padded with zeros)
__global__ void pack_w0T_kernel(const float* __restrict__ w0s, _Float16* __restrict__ w0T) {
    int idx = blockIdx.x * blockDim.x + threadIdx.x;     // S*8*64*96
    int k = idx % K0PAD;
    int o = (idx / K0PAD) % F0OUT;
    int sh = idx / (K0PAD * F0OUT);                       // s*8+h
    float v = (k < 66) ? w0s[(sh * 66 + k) * F0OUT + o] : 0.0f;
    w0T[idx] = (_Float16)v;
}

// pack w1 [S,512,16] f32 -> w1T [S,16,512] f16
__global__ void pack_w1T_kernel(const float* __restrict__ w1s, _Float16* __restrict__ w1T) {
    int idx = blockIdx.x * blockDim.x + threadIdx.x;     // S*16*512
    int k = idx % F1IN;
    int o = (idx / F1IN) % F1OUT;
    int s = idx / (F1IN * F1OUT);
    w1T[idx] = (_Float16)w1s[(s * F1IN + k) * F1OUT + o];
}

// x0 [B,N,96] f16 = concat(x[B,N,2], emb[B,N,64]) zero-padded
__global__ void build_x0_kernel(const float* __restrict__ x, const float* __restrict__ emb,
                                _Float16* __restrict__ x0) {
    int idx = blockIdx.x * blockDim.x + threadIdx.x;     // B*N*96
    int k = idx % K0PAD;
    int bn = idx / K0PAD;                                 // b*256+n
    float v = 0.0f;
    if (k < 2)       v = x[bn * 2 + k];
    else if (k < 66) v = emb[bn * 64 + (k - 2)];
    x0[idx] = (_Float16)v;
}

// ------------------------ Layer 0: projection GEMM ---------------------------
// h0T[s,b,h][o][m] = sum_k x0[b,m,k] * w0[s,h,k,o]  (stored transposed, f16)
__global__ __launch_bounds__(256) void gemm0_kernel(const _Float16* __restrict__ x0,
                                                    const _Float16* __restrict__ w0T,
                                                    _Float16* __restrict__ h0T) {
    int lane = threadIdx.x & 31;
    int wave = (blockIdx.x * blockDim.x + threadIdx.x) >> 5;  // S*B*16*32 = 32768
    int sb = wave >> 9;
    int t  = wave & 511;
    int mt = t >> 5;            // 16 M tiles (nodes)
    int nt = t & 31;            // 32 N tiles (h*64+o)
    int s = sb >> 4, b = sb & 15;
    int h = nt >> 2;
    int o0 = (nt & 3) << 4;
    const _Float16* A  = x0  + (size_t)(b * NN + mt * 16) * K0PAD;
    const _Float16* Bt = w0T + (size_t)((s * NH0 + h) * F0OUT + o0) * K0PAD;
    v8f c = {};
#pragma unroll
    for (int kb = 0; kb < K0PAD; kb += 32) {
        v16h a  = load_a16(A + kb, K0PAD, lane);
        v16h bb = load_bt16(Bt + kb, K0PAD, lane);
        c = __builtin_amdgcn_wmma_f32_16x16x32_f16(false, a, false, bb, (short)0, c, false, false);
    }
    // transposed store: D[o][node]
    _Float16* D = h0T + (size_t)(((s * BATCH + b) * NH0 + h) * F0OUT) * NN;
    int n_loc = lane & 15, hf = lane >> 4;
#pragma unroll
    for (int v = 0; v < 8; ++v) {
        int node = mt * 16 + v + 8 * hf;
        D[(size_t)(o0 + n_loc) * NN + node] = (_Float16)c[v];
    }
}

// e_src/e_dst = sum_o tanh(h0) * a  ; one thread per (s,b,h,n)
__global__ void e0_kernel(const _Float16* __restrict__ h0T,
                          const float* __restrict__ a0s, const float* __restrict__ a0d,
                          float* __restrict__ e0s, float* __restrict__ e0d) {
    int idx = blockIdx.x * blockDim.x + threadIdx.x;   // ((s*B+b)*H0+h)*N+n
    int n = idx & 255;
    int h = (idx >> 8) & 7;
    int b = (idx >> 11) & 15;
    int s = idx >> 15;
    const _Float16* base = h0T + (size_t)(((s * BATCH + b) * NH0 + h) * F0OUT) * NN + n;
    const float* as = a0s + (s * NH0 + h) * F0OUT;
    const float* ad = a0d + (s * NH0 + h) * F0OUT;
    float es = 0.0f, ed = 0.0f;
    for (int o = 0; o < F0OUT; ++o) {
        float t = tanhf((float)base[(size_t)o * NN]);
        es += t * as[o];
        ed += t * ad[o];
    }
    e0s[idx] = es; e0d[idx] = ed;
}

// fused: per (b,h,n) row, softmax over m for each of S samples, then var/mean filter
__global__ __launch_bounds__(256) void attn0_kernel(const int* __restrict__ adj,
                                                    const float* __restrict__ e0s,
                                                    const float* __restrict__ e0d,
                                                    _Float16* __restrict__ attf0) {
    int lane = threadIdx.x & 31;
    int row = (blockIdx.x * blockDim.x + threadIdx.x) >> 5;  // B*H0*N = 32768 rows
    int n = row & 255;
    int h = (row >> 8) & 7;
    int b = row >> 11;
    const int* maskRow = adj + (size_t)(b * NN + n) * NN;
    bool mvalid[8];
#pragma unroll
    for (int j = 0; j < 8; ++j) mvalid[j] = maskRow[lane + 32 * j] > 0;
    float m1[8] = {0,0,0,0,0,0,0,0};
    float m2[8] = {0,0,0,0,0,0,0,0};
    for (int s = 0; s < SAMP; ++s) {
        int ebase = ((s * BATCH + b) * NH0 + h) * NN;
        float es = e0s[ebase + n];
        float l[8];
        float mx = NEGINF;
#pragma unroll
        for (int j = 0; j < 8; ++j) {
            float v = es + e0d[ebase + lane + 32 * j];
            v = (v > 0.0f) ? v : 0.2f * v;               // leaky_relu(0.2)
            l[j] = mvalid[j] ? v : NEGINF;
            mx = fmaxf(mx, l[j]);
        }
#pragma unroll
        for (int off = 16; off > 0; off >>= 1) mx = fmaxf(mx, __shfl_xor(mx, off, 32));
        float sum = 0.0f;
#pragma unroll
        for (int j = 0; j < 8; ++j) { l[j] = __expf(l[j] - mx); sum += l[j]; }
#pragma unroll
        for (int off = 16; off > 0; off >>= 1) sum += __shfl_xor(sum, off, 32);
        float inv = 1.0f / sum;
#pragma unroll
        for (int j = 0; j < 8; ++j) { float a = l[j] * inv; m1[j] += a; m2[j] += a * a; }
    }
    _Float16* out = attf0 + (size_t)((b * NH0 + h) * NN + n) * NN;
#pragma unroll
    for (int j = 0; j < 8; ++j) {
        float mean = m1[j] * 0.25f;
        float var  = m2[j] * 0.25f - mean * mean;        // unbiased=False
        float ratio = var / (mean + 1e-6f);
        out[lane + 32 * j] = (_Float16)((ratio > A_LIMIT) ? A_FILTERED : mean);
    }
}

// x1[s,b,n, h*64+o] = ELU( sum_m attf0[b,h,n,m]*h0[s,b,h,m,o] + b0[s,o] )
__global__ __launch_bounds__(256) void av0_kernel(const _Float16* __restrict__ attf0,
                                                  const _Float16* __restrict__ h0T,
                                                  const float* __restrict__ b0s,
                                                  _Float16* __restrict__ x1) {
    int lane = threadIdx.x & 31;
    int wave = (blockIdx.x * blockDim.x + threadIdx.x) >> 5;  // S*B*H0*64 = 32768
    int t = wave & 63;
    int sbh = wave >> 6;
    int mt = t >> 2, nt = t & 3;
    int h = sbh & 7;
    int b = (sbh >> 3) & 15;
    int s = sbh >> 7;
    const _Float16* A  = attf0 + (size_t)((b * NH0 + h) * NN + mt * 16) * NN;
    const _Float16* Bt = h0T + (size_t)(((s * BATCH + b) * NH0 + h) * F0OUT + nt * 16) * NN;
    v8f c = {};
#pragma unroll
    for (int kb = 0; kb < NN; kb += 32) {
        v16h a  = load_a16(A + kb, NN, lane);
        v16h bb = load_bt16(Bt + kb, NN, lane);
        c = __builtin_amdgcn_wmma_f32_16x16x32_f16(false, a, false, bb, (short)0, c, false, false);
    }
    int n_loc = lane & 15, hf = lane >> 4;
    int o = nt * 16 + n_loc;
    float bias = b0s[s * F0OUT + o];
    _Float16* out = x1 + (size_t)((s * BATCH + b) * NN) * F1IN;
#pragma unroll
    for (int v = 0; v < 8; ++v) {
        int node = mt * 16 + v + 8 * hf;
        float val = c[v] + bias;
        val = (val > 0.0f) ? val : expm1f(val);           // ELU
        out[(size_t)node * F1IN + h * F0OUT + o] = (_Float16)val;
    }
}

// ------------------------ Layer 1: projection GEMM ---------------------------
__global__ __launch_bounds__(256) void gemm1_kernel(const _Float16* __restrict__ x1,
                                                    const _Float16* __restrict__ w1T,
                                                    float* __restrict__ h1) {
    int lane = threadIdx.x & 31;
    int wave = (blockIdx.x * blockDim.x + threadIdx.x) >> 5;  // S*B*16 = 1024
    int mt = wave & 15;
    int sb = wave >> 4;
    int s = sb >> 4, b = sb & 15;
    const _Float16* A  = x1  + (size_t)((s * BATCH + b) * NN + mt * 16) * F1IN;
    const _Float16* Bt = w1T + (size_t)s * F1OUT * F1IN;
    v8f c = {};
#pragma unroll 4
    for (int kb = 0; kb < F1IN; kb += 32) {
        v16h a  = load_a16(A + kb, F1IN, lane);
        v16h bb = load_bt16(Bt + kb, F1IN, lane);
        c = __builtin_amdgcn_wmma_f32_16x16x32_f16(false, a, false, bb, (short)0, c, false, false);
    }
    float* D = h1 + (size_t)((s * BATCH + b) * NN + mt * 16) * F1OUT;
    int n_loc = lane & 15, hf = lane >> 4;
#pragma unroll
    for (int v = 0; v < 8; ++v) D[(size_t)(v + 8 * hf) * F1OUT + n_loc] = c[v];
}

__global__ void e1_kernel(const float* __restrict__ h1,
                          const float* __restrict__ a1s, const float* __restrict__ a1d,
                          float* __restrict__ e1s, float* __restrict__ e1d) {
    int idx = blockIdx.x * blockDim.x + threadIdx.x;   // S*B*N = 16384
    const float* base = h1 + (size_t)idx * F1OUT;
    int s = idx / (BATCH * NN);
    float es = 0.0f, ed = 0.0f;
    for (int o = 0; o < F1OUT; ++o) {
        float t = tanhf(base[o]);
        es += t * a1s[s * F1OUT + o];
        ed += t * a1d[s * F1OUT + o];
    }
    e1s[idx] = es; e1d[idx] = ed;
}

// only row n = N-1 is ever read by the output -> one filtered attention row per batch
__global__ __launch_bounds__(256) void attn1_kernel(const int* __restrict__ adj,
                                                    const float* __restrict__ e1s,
                                                    const float* __restrict__ e1d,
                                                    float* __restrict__ attf1) {
    __shared__ float red[256];
    int b = blockIdx.x;
    int m = threadIdx.x;
    bool valid = adj[((size_t)(b * NN) + (NN - 1)) * NN + m] > 0;
    float m1 = 0.0f, m2 = 0.0f;
    for (int s = 0; s < SAMP; ++s) {
        int ebase = (s * BATCH + b) * NN;
        float v = e1s[ebase + (NN - 1)] + e1d[ebase + m];
        v = (v > 0.0f) ? v : 0.2f * v;
        float l = valid ? v : NEGINF;
        red[m] = l; __syncthreads();
        for (int off = 128; off > 0; off >>= 1) {
            if (m < off) red[m] = fmaxf(red[m], red[m + off]);
            __syncthreads();
        }
        float mx = red[0]; __syncthreads();
        float e = __expf(l - mx);
        red[m] = e; __syncthreads();
        for (int off = 128; off > 0; off >>= 1) {
            if (m < off) red[m] += red[m + off];
            __syncthreads();
        }
        float a = e / red[0]; __syncthreads();
        m1 += a; m2 += a * a;
    }
    float mean = m1 * 0.25f;
    float var  = m2 * 0.25f - mean * mean;
    float ratio = var / (mean + 1e-6f);
    attf1[b * NN + m] = (ratio > A_LIMIT) ? A_FILTERED : mean;
}

// out[b,o] = mean_s log_softmax_o( sum_m attf1[b,m]*h1[s,b,m,o] + b1[s,o] )
__global__ __launch_bounds__(64) void final_kernel(const float* __restrict__ attf1,
                                                   const float* __restrict__ h1,
                                                   const float* __restrict__ b1s,
                                                   float* __restrict__ out) {
    __shared__ float vals[SAMP][F1OUT];
    int b = blockIdx.x;
    int tid = threadIdx.x;               // 64 threads: s = tid>>4, o = tid&15
    int s = tid >> 4, o = tid & 15;
    const float* hb = h1 + (size_t)(s * BATCH + b) * NN * F1OUT;
    const float* af = attf1 + b * NN;
    float acc = 0.0f;
    for (int m = 0; m < NN; ++m) acc += af[m] * hb[m * F1OUT + o];
    vals[s][o] = acc + b1s[s * F1OUT + o];
    __syncthreads();
    if (tid < F1OUT) {
        float r = 0.0f;
        for (int ss = 0; ss < SAMP; ++ss) {
            float mx = -1e30f;
            for (int k = 0; k < F1OUT; ++k) mx = fmaxf(mx, vals[ss][k]);
            float sum = 0.0f;
            for (int k = 0; k < F1OUT; ++k) sum += __expf(vals[ss][k] - mx);
            r += vals[ss][tid] - mx - logf(sum);
        }
        out[b * F1OUT + tid] = r * 0.25f;
    }
}

// ---------------------------------------------------------------------------
extern "C" void kernel_launch(void* const* d_in, const int* in_sizes, int n_in,
                              void* d_out, int out_size, void* d_ws, size_t ws_size,
                              hipStream_t stream) {
    const int*   adj = (const int*)d_in[0];
    const float* x   = (const float*)d_in[1];
    const float* emb = (const float*)d_in[2];

    size_t off = 0;
    auto alloc = [&](size_t bytes) -> void* {
        void* p = (char*)d_ws + off;
        off += (bytes + 255) & ~(size_t)255;
        return p;
    };

    float* w0s  = (float*)alloc((size_t)SAMP * NH0 * 66 * F0OUT * 4);
    float* a0s_ = (float*)alloc((size_t)SAMP * NH0 * F0OUT * 4);
    float* a0d_ = (float*)alloc((size_t)SAMP * NH0 * F0OUT * 4);
    float* b0s  = (float*)alloc((size_t)SAMP * F0OUT * 4);
    float* w1s  = (float*)alloc((size_t)SAMP * F1IN * F1OUT * 4);
    float* a1s_ = (float*)alloc((size_t)SAMP * F1OUT * 4);
    float* a1d_ = (float*)alloc((size_t)SAMP * F1OUT * 4);
    float* b1s  = (float*)alloc((size_t)SAMP * F1OUT * 4);
    _Float16* w0T   = (_Float16*)alloc((size_t)SAMP * NH0 * F0OUT * K0PAD * 2);
    _Float16* w1T   = (_Float16*)alloc((size_t)SAMP * F1OUT * F1IN * 2);
    _Float16* x0    = (_Float16*)alloc((size_t)BATCH * NN * K0PAD * 2);
    _Float16* h0T   = (_Float16*)alloc((size_t)SAMP * BATCH * NH0 * F0OUT * NN * 2);
    float*    e0s   = (float*)alloc((size_t)SAMP * BATCH * NH0 * NN * 4);
    float*    e0d   = (float*)alloc((size_t)SAMP * BATCH * NH0 * NN * 4);
    _Float16* attf0 = (_Float16*)alloc((size_t)BATCH * NH0 * NN * NN * 2);
    _Float16* x1    = (_Float16*)alloc((size_t)SAMP * BATCH * NN * F1IN * 2);
    float*    h1    = (float*)alloc((size_t)SAMP * BATCH * NN * F1OUT * 4);
    float*    e1s   = (float*)alloc((size_t)SAMP * BATCH * NN * 4);
    float*    e1d   = (float*)alloc((size_t)SAMP * BATCH * NN * 4);
    float*    attf1 = (float*)alloc((size_t)BATCH * NN * 4);
    if (off > ws_size) return;  // workspace too small (needs ~56 MB)

    auto drawN = [&](int muIdx, float* out, int nPer, int fold) {
        int total = SAMP * nPer;
        draw_kernel<<<(total + 255) / 256, 256, 0, stream>>>(
            (const float*)d_in[muIdx], (const float*)d_in[muIdx + 1], out, nPer, total, fold);
    };
    drawN(3,  w0s,  NH0 * 66 * F0OUT, 0);   // w0
    drawN(5,  a0s_, NH0 * F0OUT,      1);   // a_src0
    drawN(7,  a0d_, NH0 * F0OUT,      2);   // a_dst0
    drawN(9,  b0s,  F0OUT,            3);   // b0
    drawN(11, w1s,  F1IN * F1OUT,     4);   // w1
    drawN(13, a1s_, F1OUT,            5);   // a_src1
    drawN(15, a1d_, F1OUT,            6);   // a_dst1
    drawN(17, b1s,  F1OUT,            7);   // b1

    pack_w0T_kernel<<<(SAMP * NH0 * F0OUT * K0PAD) / 256, 256, 0, stream>>>(w0s, w0T);
    pack_w1T_kernel<<<(SAMP * F1OUT * F1IN) / 256, 256, 0, stream>>>(w1s, w1T);
    build_x0_kernel<<<(BATCH * NN * K0PAD) / 256, 256, 0, stream>>>(x, emb, x0);

    gemm0_kernel<<<4096, 256, 0, stream>>>(x0, w0T, h0T);              // 32768 waves
    e0_kernel<<<(SAMP * BATCH * NH0 * NN) / 256, 256, 0, stream>>>(h0T, a0s_, a0d_, e0s, e0d);
    attn0_kernel<<<4096, 256, 0, stream>>>(adj, e0s, e0d, attf0);      // one wave/row
    av0_kernel<<<4096, 256, 0, stream>>>(attf0, h0T, b0s, x1);         // 32768 waves
    gemm1_kernel<<<128, 256, 0, stream>>>(x1, w1T, h1);                // 1024 waves
    e1_kernel<<<(SAMP * BATCH * NN) / 256, 256, 0, stream>>>(h1, a1s_, a1d_, e1s, e1d);
    attn1_kernel<<<BATCH, 256, 0, stream>>>(adj, e1s, e1d, attf1);
    final_kernel<<<BATCH, 64, 0, stream>>>(attf1, h1, b1s, (float*)d_out);
}
